// CausalSelectiveSelfAttentionWithMemoryPenalty_24713241821845
// MI455X (gfx1250) — matrix operand order repaired and checked by reading, not compiled, not run on hardware
//
#include <hip/hip_runtime.h>
#include <stdint.h>

// CDNA5 (gfx1250) wave32 WMMA implementation of causal self-attention with
// memory penalty.  All GEMMs (QKV proj, QK^T, PV, out proj) run on
// v_wmma_f32_16x16x32_bf16 with f32 accumulation; softmax / FF penalty stay f32.

typedef __bf16 bf16_t;
typedef __attribute__((ext_vector_type(16))) __bf16 v16bf;
typedef __attribute__((ext_vector_type(8)))  __bf16 v8bf;
typedef __attribute__((ext_vector_type(8)))  float   v8f;
typedef __attribute__((ext_vector_type(4)))  float   v4f;

#define DEVI __device__ __forceinline__

namespace {
constexpr int Bn  = 2;
constexpr int Tn  = 2048;
constexpr int Cn  = 768;
constexpr int NHn = 12;
constexpr int HDn = 64;
constexpr int BTn = Bn * Tn;     // 4096
constexpr int N3n = 3 * Cn;      // 2304
constexpr float SCALE = 0.125f;  // 1/sqrt(64)
constexpr float NEGF  = -3.0e38f;
} // namespace

union U16bf { v16bf v; v8bf h[2]; };
union U8bf  { v8bf v; bf16_t e[8]; };

DEVI bf16_t f2bf(float f) {
  uint32_t u = __float_as_uint(f);
  u += 0x7FFFu + ((u >> 16) & 1u);         // round-to-nearest-even
  uint16_t s16 = (uint16_t)(u >> 16);
  bf16_t r;
  __builtin_memcpy(&r, &s16, 2);
  return r;
}

// A fragment: 16(M) x 32(K) bf16, row-major source, row stride `ld` elements.
// ISA layout: lane L<16 -> row L, elems 0..7 = K0..7, elems 8..15 = K16..23;
//             lane L+16 -> row L, elems 0..7 = K8..15, elems 8..15 = K24..31.
DEVI v16bf load_a16x32(const bf16_t* p, int ld) {
  const int lane = threadIdx.x & 31;
  const int row  = lane & 15;
  const int hi   = lane >> 4;
  const bf16_t* q = p + (size_t)row * ld + hi * 8;
  U16bf u;
  u.h[0] = *(const v8bf*)(q);
  u.h[1] = *(const v8bf*)(q + 16);
  return u.v;
}

// B fragment: 32(K) x 16(N) bf16 from a [N][K]-contiguous source (per-column
// contiguous K), column stride `ld`.  lanes 0-15: col=lane, K=0..7,16..23;
// lanes 16-31: col=lane-16, K=8..15,24..31 (same K order as A).
DEVI v16bf load_b32x16(const bf16_t* p, int ld) {
  const int lane = threadIdx.x & 31;
  const int col  = lane & 15;
  const int seg  = lane >> 4;
  const bf16_t* q = p + (size_t)col * ld + seg * 8;
  U16bf u;
  u.h[0] = *(const v8bf*)(q);
  u.h[1] = *(const v8bf*)(q + 16);
  return u.v;
}

DEVI v8f wmma_bf16(v16bf a, v16bf b, v8f c) {
  return __builtin_amdgcn_wmma_f32_16x16x32_bf16(false, a, false, b, (short)0, c,
                                                 false, false);
}

// ---------------------------------------------------------------- converters
__global__ __launch_bounds__(256) void k_cvt_bf16(const float* __restrict__ in,
                                                  bf16_t* __restrict__ out, int n) {
  int i = blockIdx.x * 256 + threadIdx.x;
  if (i < n) out[i] = f2bf(in[i]);
}

// out[n*K+k] = in[k*N+n]  (weight transpose -> [N][K] bf16)
__global__ __launch_bounds__(256) void k_cvt_transpose(const float* __restrict__ in,
                                                       bf16_t* __restrict__ out,
                                                       int N, int K) {
  int i = blockIdx.x * 256 + threadIdx.x;
  if (i < N * K) {
    int n = i / K, k = i - n * K;
    out[i] = f2bf(in[(size_t)k * N + n]);
  }
}

// --------------------------------------------------------------- QKV GEMM
// xb[BT][C] @ waT -> qkv[BT][2304]; scatter to Q,K ([b][h][t][d]) and
// V transposed ([b][h][d][t]).  One wave computes a 32x64 output tile
// (2 A fragments reuse each B fragment -> ~1.8x better L2 intensity).
__global__ __launch_bounds__(256) void k_qkv_gemm(const bf16_t* __restrict__ xb,
                                                  const bf16_t* __restrict__ waT,
                                                  const float*  __restrict__ bias,
                                                  bf16_t* __restrict__ Q,
                                                  bf16_t* __restrict__ K,
                                                  bf16_t* __restrict__ Vt) {
  const int lane = threadIdx.x & 31;
  const int wid  = (blockIdx.x * 256 + threadIdx.x) >> 5;   // 0..4607
  const int nt = wid % (N3n / 64);
  const int mt = wid / (N3n / 64);
  const int m0 = mt * 32;
  const int n0 = nt * 64;

  v8f acc[2][4] = {};
  for (int k0 = 0; k0 < Cn; k0 += 32) {
    // prefetch next K-chunk of the A panel (global_prefetch_b8, no counter)
    if (k0 + 32 < Cn)
      __builtin_prefetch(xb + (size_t)(m0 + (threadIdx.x & 15)) * Cn + k0 + 32, 0, 1);
    v16bf a0 = load_a16x32(xb + (size_t)m0 * Cn + k0, Cn);
    v16bf a1 = load_a16x32(xb + (size_t)(m0 + 16) * Cn + k0, Cn);
#pragma unroll
    for (int s = 0; s < 4; ++s) {
      v16bf bf = load_b32x16(waT + (size_t)(n0 + s * 16) * Cn + k0, Cn);
      acc[0][s] = wmma_bf16(a0, bf, acc[0][s]);
      acc[1][s] = wmma_bf16(a1, bf, acc[1][s]);
    }
  }

  const int sel = nt / NHn;            // 0=q 1=k 2=v
  const int h   = nt % NHn;
  const int hi  = lane >> 4;
  const int cl  = lane & 15;
  const int b   = m0 >> 11;            // tile never crosses batch (2048%32==0)

#pragma unroll
  for (int u = 0; u < 2; ++u) {
    const int t0 = (m0 & (Tn - 1)) + u * 16 + hi * 8;
#pragma unroll
    for (int s = 0; s < 4; ++s) {
      const int dd = s * 16 + cl;
      const float bv = bias[n0 + s * 16 + cl];
      if (sel < 2) {
        bf16_t* dst = (sel == 0 ? Q : K) + ((size_t)(b * NHn + h) * Tn) * HDn;
#pragma unroll
        for (int r = 0; r < 8; ++r)
          dst[(size_t)(t0 + r) * HDn + dd] = f2bf(acc[u][s][r] + bv);
      } else {
        bf16_t* dst = Vt + ((size_t)(b * NHn + h) * HDn + dd) * Tn + t0;
        U8bf t8;
#pragma unroll
        for (int r = 0; r < 8; ++r) t8.e[r] = f2bf(acc[u][s][r] + bv);
        *(v8bf*)dst = t8.v;            // 8 tokens contiguous in Vt
      }
    }
  }
}

// ------------------------------------------------- head-0 masked score matrix
// S[b][r][k] = (k<r && k!=0) ? relu(q0[r].k0[k] * 0.125) : 0   (into FF buffer)
__global__ __launch_bounds__(256) void k_score0(const bf16_t* __restrict__ Q,
                                                const bf16_t* __restrict__ K,
                                                float* __restrict__ S) {
  const int lane = threadIdx.x & 31;
  const int wid  = (blockIdx.x * 256 + threadIdx.x) >> 5;   // 0..32767
  const int b   = wid >> 14;
  const int rem = wid & 16383;
  const int rt  = rem >> 7;
  const int kt  = rem & 127;
  const int r0  = rt * 16;
  const int k0  = kt * 16;
  const int hi  = lane >> 4;
  const int cl  = lane & 15;

  float* out = S + ((size_t)b * Tn) * Tn;

  if (k0 > r0 + 15) {                  // tile entirely above diagonal
#pragma unroll
    for (int r = 0; r < 8; ++r)
      out[(size_t)(r0 + hi * 8 + r) * Tn + k0 + cl] = 0.0f;
    return;
  }
  const bf16_t* Qh = Q + ((size_t)(b * NHn) * Tn) * HDn;    // head 0
  const bf16_t* Kh = K + ((size_t)(b * NHn) * Tn) * HDn;
  v8f c = {};
  c = wmma_bf16(load_a16x32(Qh + (size_t)r0 * HDn + 0, HDn),
                load_b32x16(Kh + (size_t)k0 * HDn + 0, HDn), c);
  c = wmma_bf16(load_a16x32(Qh + (size_t)r0 * HDn + 32, HDn),
                load_b32x16(Kh + (size_t)k0 * HDn + 32, HDn), c);
#pragma unroll
  for (int r = 0; r < 8; ++r) {
    const int row = r0 + hi * 8 + r;
    const int col = k0 + cl;
    float v = c[r] * SCALE;
    v = (col < row && col != 0) ? fmaxf(v, 0.0f) : 0.0f;
    out[(size_t)row * Tn + col] = v;
  }
}

// ------------------------------------- exclusive prefix over query rows (FF)
__global__ __launch_bounds__(256) void k_ff_prefix(float* __restrict__ S) {
  const int b  = blockIdx.x >> 3;
  const int k  = (blockIdx.x & 7) * 256 + threadIdx.x;
  float acc = 0.0f;
  float* base = S + ((size_t)b * Tn) * Tn + k;
  for (int i = 0; i < Tn; ++i) {
    float v = base[(size_t)i * Tn];
    base[(size_t)i * Tn] = acc;        // exclusive cumsum, in place
    acc += v;
  }
}

// FF_sum[b][i] = sum_k clip(FF,0,TAU)   (TAU = 1)
__global__ __launch_bounds__(256) void k_ff_rowsum(const float* __restrict__ FF,
                                                   float* __restrict__ FFs) {
  const int bi = blockIdx.x;           // b*T + i
  const float* row = FF + (size_t)bi * Tn;
  float s = 0.0f;
  for (int k = threadIdx.x; k < Tn; k += 256)
    s += fminf(fmaxf(row[k], 0.0f), 1.0f);
  __shared__ float red[256];
  red[threadIdx.x] = s;
  __syncthreads();
  for (int off = 128; off > 0; off >>= 1) {
    if (threadIdx.x < off) red[threadIdx.x] += red[threadIdx.x + off];
    __syncthreads();
  }
  if (threadIdx.x == 0) FFs[bi] = red[0];
}

// M[0,b,i,j] = i - FF_sum[b][j]
__global__ __launch_bounds__(256) void k_m_write(const float* __restrict__ FFs,
                                                 float* __restrict__ M) {
  size_t idx = (size_t)blockIdx.x * 256 + threadIdx.x;      // < B*T*T
  int j = (int)(idx & (Tn - 1));
  int i = (int)((idx >> 11) & (Tn - 1));
  int b = (int)(idx >> 22);
  M[idx] = (float)i - FFs[b * Tn + j];
}

// ---------------------------------------------- flash-softmax tile update
// c0/c1 hold S^T tiles (keys kb..kb+15 / kb+16..kb+31 in rows, queries in
// lanes).  Returns exp() probabilities packed in the A-fragment layout for PV.
DEVI v16bf softmax_tile(v8f c0, v8f c1, const float* FFrow, int kb, int qg,
                        int hi, float& m_run, float& l_run, float& alpha_out) {
  const int kbase = kb + hi * 8;       // this lane's first key row
  v4f f0a = *(const v4f*)(FFrow + kbase);
  v4f f0b = *(const v4f*)(FFrow + kbase + 4);
  v4f f1a = *(const v4f*)(FFrow + kbase + 16);
  v4f f1b = *(const v4f*)(FFrow + kbase + 20);
  float s0[8], s1[8];
  float mx = NEGF;
#pragma unroll
  for (int r = 0; r < 8; ++r) {
    const int key0 = kbase + r;
    float ff0 = (r < 4) ? f0a[r] : f0b[r - 4];
    float v0  = c0[r] * SCALE - ff0;
    s0[r] = (key0 <= qg) ? v0 : NEGF;
    float ff1 = (r < 4) ? f1a[r] : f1b[r - 4];
    float v1  = c1[r] * SCALE - ff1;
    s1[r] = (key0 + 16 <= qg) ? v1 : NEGF;
    mx = fmaxf(mx, fmaxf(s0[r], s1[r]));
  }
  mx = fmaxf(mx, __shfl_xor(mx, 16));              // combine key halves
  const float mnew  = fmaxf(m_run, mx);
  const float alpha = __expf(m_run - mnew);
  float rsum = 0.0f;
  U16bf p;
#pragma unroll
  for (int r = 0; r < 8; ++r) {
    float p0 = __expf(s0[r] - mnew);
    float p1 = __expf(s1[r] - mnew);
    rsum += p0 + p1;
    p.v[r]     = f2bf(p0);                         // keys kb+hi*8+r
    p.v[r + 8] = f2bf(p1);                         // keys kb+16+hi*8+r
  }
  rsum += __shfl_xor(rsum, 16);
  l_run = l_run * alpha + rsum;
  m_run = mnew;
  alpha_out = alpha;
  return p.v;
}

// ------------------------------------------------------- flash attention
// One wave per (b, h, 32-query tile): two 16-query sub-tiles share every K
// A-fragment and V B-fragment (halves K/V L2 traffic, 16 WMMA per 32 keys).
// S^T trick: A=K, B=Q^T so exp() results land directly in PV's A layout.
__global__ __launch_bounds__(256) void k_attn(const bf16_t* __restrict__ Q,
                                              const bf16_t* __restrict__ K,
                                              const bf16_t* __restrict__ Vt,
                                              const float*  __restrict__ FF,
                                              bf16_t* __restrict__ yh) {
  const int lane = threadIdx.x & 31;
  const int wid  = (blockIdx.x * 256 + threadIdx.x) >> 5;   // 0..1535
  const int qt = wid & 63;             // 64 tiles of 32 queries
  const int h  = (wid >> 6) % NHn;
  const int b  = wid / (64 * NHn);
  const int q0 = qt * 32;
  const int hi = lane >> 4;
  const int ql = lane & 15;            // this lane's query column

  const bf16_t* Qh = Q  + ((size_t)(b * NHn + h) * Tn) * HDn;
  const bf16_t* Kh = K  + ((size_t)(b * NHn + h) * Tn) * HDn;
  const bf16_t* Vh = Vt + ((size_t)(b * NHn + h) * HDn) * Tn;
  const float*  FF0 = FF + ((size_t)(b * Tn) + q0 + ql) * Tn;        // u=0
  const float*  FF1 = FF + ((size_t)(b * Tn) + q0 + 16 + ql) * Tn;   // u=1

  v16bf qb00 = load_b32x16(Qh + (size_t)(q0)      * HDn + 0,  HDn);
  v16bf qb01 = load_b32x16(Qh + (size_t)(q0)      * HDn + 32, HDn);
  v16bf qb10 = load_b32x16(Qh + (size_t)(q0 + 16) * HDn + 0,  HDn);
  v16bf qb11 = load_b32x16(Qh + (size_t)(q0 + 16) * HDn + 32, HDn);

  float m0r = NEGF, l0r = 0.0f, m1r = NEGF, l1r = 0.0f;
  v8f O0[4] = {}, O1[4] = {};

  for (int kb = 0; kb <= q0 + 31; kb += 32) {
    // K fragments for both 16-key sub-tiles, shared by both query sub-tiles.
    v16bf ka0 = load_a16x32(Kh + (size_t)kb * HDn + 0,  HDn);
    v16bf ka1 = load_a16x32(Kh + (size_t)kb * HDn + 32, HDn);
    v16bf kc0 = load_a16x32(Kh + (size_t)(kb + 16) * HDn + 0,  HDn);
    v16bf kc1 = load_a16x32(Kh + (size_t)(kb + 16) * HDn + 32, HDn);
    v8f c00 = {}, c01 = {}, c10 = {}, c11 = {};
    c00 = wmma_bf16(ka0, qb00, c00); c00 = wmma_bf16(ka1, qb01, c00);
    c01 = wmma_bf16(kc0, qb00, c01); c01 = wmma_bf16(kc1, qb01, c01);
    c10 = wmma_bf16(ka0, qb10, c10); c10 = wmma_bf16(ka1, qb11, c10);
    c11 = wmma_bf16(kc0, qb10, c11); c11 = wmma_bf16(kc1, qb11, c11);

    float a0, a1;
    v16bf p0 = softmax_tile(c00, c01, FF0, kb, q0 + ql,      hi, m0r, l0r, a0);
    v16bf p1 = softmax_tile(c10, c11, FF1, kb, q0 + 16 + ql, hi, m1r, l1r, a1);
#pragma unroll
    for (int r = 0; r < 8; ++r) {      // rescale O rows (per-query alpha)
      const int src = hi ? (8 + r) : r;
      float a0r = __shfl(a0, src, 32);
      float a1r = __shfl(a1, src, 32);
#pragma unroll
      for (int c = 0; c < 4; ++c) { O0[c][r] *= a0r; O1[c][r] *= a1r; }
    }
#pragma unroll
    for (int c = 0; c < 4; ++c) {      // V fragment shared by both sub-tiles
      v16bf vb = load_b32x16(Vh + (size_t)(c * 16) * Tn + kb, Tn);
      O0[c] = wmma_bf16(p0, vb, O0[c]);
      O1[c] = wmma_bf16(p1, vb, O1[c]);
    }
  }
#pragma unroll
  for (int r = 0; r < 8; ++r) {
    const int src = hi ? (8 + r) : r;
    float i0 = 1.0f / __shfl(l0r, src, 32);
    float i1 = 1.0f / __shfl(l1r, src, 32);
    const int t0 = q0 + hi * 8 + r;
    bf16_t* y0 = yh + ((size_t)(b * Tn) + t0)      * Cn + h * HDn;
    bf16_t* y1 = yh + ((size_t)(b * Tn) + t0 + 16) * Cn + h * HDn;
#pragma unroll
    for (int c = 0; c < 4; ++c) {
      y0[c * 16 + ql] = f2bf(O0[c][r] * i0);
      y1[c * 16 + ql] = f2bf(O1[c][r] * i1);
    }
  }
}

// ---------------------------------------------------------- output projection
// One wave computes a 32x64 tile (2 A fragments per B fragment).
__global__ __launch_bounds__(256) void k_proj_gemm(const bf16_t* __restrict__ yh,
                                                   const bf16_t* __restrict__ wpT,
                                                   const float* __restrict__ bias,
                                                   float* __restrict__ out) {
  const int lane = threadIdx.x & 31;
  const int wid  = (blockIdx.x * 256 + threadIdx.x) >> 5;   // 0..1535
  const int nt = wid % (Cn / 64);
  const int mt = wid / (Cn / 64);
  const int m0 = mt * 32;
  const int n0 = nt * 64;
  v8f acc[2][4] = {};
  for (int k0 = 0; k0 < Cn; k0 += 32) {
    if (k0 + 32 < Cn)
      __builtin_prefetch(yh + (size_t)(m0 + (threadIdx.x & 15)) * Cn + k0 + 32, 0, 1);
    v16bf a0 = load_a16x32(yh + (size_t)m0 * Cn + k0, Cn);
    v16bf a1 = load_a16x32(yh + (size_t)(m0 + 16) * Cn + k0, Cn);
#pragma unroll
    for (int s = 0; s < 4; ++s) {
      v16bf bf = load_b32x16(wpT + (size_t)(n0 + s * 16) * Cn + k0, Cn);
      acc[0][s] = wmma_bf16(a0, bf, acc[0][s]);
      acc[1][s] = wmma_bf16(a1, bf, acc[1][s]);
    }
  }
  const int hi = lane >> 4;
  const int cl = lane & 15;
#pragma unroll
  for (int u = 0; u < 2; ++u) {
#pragma unroll
    for (int s = 0; s < 4; ++s) {
      const int col = n0 + s * 16 + cl;
      const float bv = bias[col];
#pragma unroll
      for (int r = 0; r < 8; ++r) {
        const int row = m0 + u * 16 + hi * 8 + r;
        out[(size_t)row * Cn + col] = acc[u][s][r] + bv;
      }
    }
  }
}

// ------------------------------------------------------------------- launch
extern "C" void kernel_launch(void* const* d_in, const int* in_sizes, int n_in,
                              void* d_out, int out_size, void* d_ws, size_t ws_size,
                              hipStream_t stream) {
  (void)in_sizes; (void)n_in; (void)out_size;
  const float* x      = (const float*)d_in[0];   // [B,T,C]
  const float* w_attn = (const float*)d_in[1];   // [C, 3C]
  const float* b_attn = (const float*)d_in[2];   // [3C]
  const float* w_proj = (const float*)d_in[3];   // [C, C]
  const float* b_proj = (const float*)d_in[4];   // [C]
  float* y_out = (float*)d_out;                          // [B,T,C]
  float* M_out = (float*)d_out + (size_t)BTn * Cn;       // [1,B,T,T]

  char* ws = (char*)d_ws;
  size_t off = 0;
  auto alloc = [&](size_t bytes) {
    char* p = ws + off;
    off += (bytes + 255) & ~(size_t)255;
    return p;
  };
  bf16_t* xb  = (bf16_t*)alloc((size_t)BTn * Cn * 2);          // 6.3 MB
  bf16_t* waT = (bf16_t*)alloc((size_t)N3n * Cn * 2);          // 3.5 MB
  bf16_t* wpT = (bf16_t*)alloc((size_t)Cn * Cn * 2);           // 1.2 MB
  bf16_t* Qb  = (bf16_t*)alloc((size_t)Bn * NHn * Tn * HDn * 2); // 6.3 MB
  bf16_t* Kb  = (bf16_t*)alloc((size_t)Bn * NHn * Tn * HDn * 2); // 6.3 MB
  bf16_t* Vt  = (bf16_t*)alloc((size_t)Bn * NHn * HDn * Tn * 2); // 6.3 MB
  float*  FF  = (float*) alloc((size_t)Bn * Tn * Tn * 4);      // 33.6 MB
  float*  FFs = (float*) alloc((size_t)Bn * Tn * 4);           // 16 KB
  bf16_t* yh  = (bf16_t*)alloc((size_t)BTn * Cn * 2);          // 6.3 MB
  if (ws_size < off) return;   // ~70 MB needed

  k_cvt_bf16<<<(BTn * Cn) / 256, 256, 0, stream>>>(x, xb, BTn * Cn);
  k_cvt_transpose<<<(N3n * Cn) / 256, 256, 0, stream>>>(w_attn, waT, N3n, Cn);
  k_cvt_transpose<<<(Cn * Cn) / 256, 256, 0, stream>>>(w_proj, wpT, Cn, Cn);

  k_qkv_gemm<<<(BTn / 32) * (N3n / 64) / 8, 256, 0, stream>>>(xb, waT, b_attn,
                                                              Qb, Kb, Vt);
  k_score0<<<(Bn * (Tn / 16) * (Tn / 16)) / 8, 256, 0, stream>>>(Qb, Kb, FF);
  k_ff_prefix<<<Bn * (Tn / 256), 256, 0, stream>>>(FF);
  k_ff_rowsum<<<Bn * Tn, 256, 0, stream>>>(FF, FFs);
  k_m_write<<<(Bn * Tn * Tn) / 256, 256, 0, stream>>>(FFs, M_out);

  k_attn<<<(Bn * NHn * (Tn / 32)) / 8, 256, 0, stream>>>(Qb, Kb, Vt, FF, yh);
  k_proj_gemm<<<(BTn / 32) * (Cn / 64) / 8, 256, 0, stream>>>(yh, wpT, b_proj,
                                                              y_out);
}
